// GQAAttention_40699110097473
// MI455X (gfx1250) — matrix operand order repaired and checked by reading, not compiled, not run on hardware
//
#include <hip/hip_runtime.h>
#include <hip/hip_bf16.h>
#include <math.h>

// ---------------------------------------------------------------------------
// CDNA5 (gfx1250) GQA attention block, WMMA bf16 path + async-LDS staging.
//   T=2048, D_MODEL=2048, N_Q=32, N_KV=8, HEAD_DIM=64, causal.
// ---------------------------------------------------------------------------

typedef __attribute__((ext_vector_type(16))) __bf16 v16bf;
typedef __attribute__((ext_vector_type(8)))  float  v8f;

#define T_CTX   2048
#define DMODEL  2048
#define NQH     32
#define NKVH    8
#define HD      64
#define LOG2E   1.44269504088896f

static __device__ inline v8f wmma_bf16(v16bf a, v16bf b, v8f c) {
    return __builtin_amdgcn_wmma_f32_16x16x32_bf16(
        false, a, false, b, (short)0, c, false, false);
}

// ---------------------------------------------------------------------------
// f32 -> bf16 bulk conversion (one pass; removes cvt from GEMM hot loops and
// halves HBM traffic for all weight/activation reads).
// ---------------------------------------------------------------------------
__global__ __launch_bounds__(256)
void cvt_bf16_kernel(const float* __restrict__ src, __bf16* __restrict__ dst, int n4)
{
    int i = blockIdx.x * 256 + threadIdx.x;
    if (i >= n4) return;
    float4 v = ((const float4*)src)[i];
    __bf16* d = dst + (size_t)i * 4;
    d[0] = (__bf16)v.x; d[1] = (__bf16)v.y;
    d[2] = (__bf16)v.z; d[3] = (__bf16)v.w;
}

// ---------------------------------------------------------------------------
// Tiled WMMA GEMM (all-bf16 inputs): C[M,N] = A[M,K]*B[K,N] + bias[N]
//   Block = 256 threads (8 waves), tile 128x128, K-step 32.
//   LDS staging via GLOBAL_LOAD_ASYNC_TO_LDS_B128 (ASYNCcnt path).
//   Wave w owns a 32x64 subtile: 2 A-frags x 4 B-frags = 8 WMMAs / K-step.
// ---------------------------------------------------------------------------
template<bool OUT_F32>
__global__ __launch_bounds__(256)
void gemm_bf16_wmma(const __bf16* __restrict__ A, const __bf16* __restrict__ B,
                    const float* __restrict__ bias, void* __restrict__ Cptr,
                    int M, int N, int K)
{
    constexpr int BM = 128, BN = 128, BK = 32;
    __shared__ __bf16 As[BM][BK];     // 8 KB
    __shared__ __bf16 Bs[BK][BN];     // 8 KB

    const int m0   = blockIdx.y * BM;
    const int n0   = blockIdx.x * BN;
    const int tid  = threadIdx.x;
    const int lane = tid & 31;
    const int wave = tid >> 5;
    const int wm   = (wave >> 1) * 32;    // 4 wave-rows
    const int wn   = (wave & 1)  * 64;    // 2 wave-cols
    const int half = lane >> 4;
    const int l16  = lane & 15;

    const unsigned as_base = (unsigned)(uintptr_t)&As[0][0];
    const unsigned bs_base = (unsigned)(uintptr_t)&Bs[0][0];

    v8f acc[2][4];
    #pragma unroll
    for (int i = 0; i < 2; ++i)
        #pragma unroll
        for (int j = 0; j < 4; ++j)
            acc[i][j] = (v8f)0.0f;

    for (int k0 = 0; k0 < K; k0 += BK) {
        // ---- async stage A tile (128x32 bf16, 16B chunks) ----
        #pragma unroll
        for (int i = 0; i < 2; ++i) {
            int l  = tid + i * 256;
            int r  = l >> 2, cc = (l & 3) * 8;
            unsigned long long ga =
                (unsigned long long)(uintptr_t)(A + (size_t)(m0 + r) * K + (k0 + cc));
            unsigned lo = as_base + (unsigned)(r * BK + cc) * 2u;
            asm volatile("global_load_async_to_lds_b128 %0, %1, off"
                         :: "v"(lo), "v"(ga) : "memory");
        }
        // ---- async stage B tile (32x128 bf16, 16B chunks) ----
        #pragma unroll
        for (int i = 0; i < 2; ++i) {
            int l  = tid + i * 256;
            int r  = l >> 4, cc = (l & 15) * 8;
            unsigned long long ga =
                (unsigned long long)(uintptr_t)(B + (size_t)(k0 + r) * N + (n0 + cc));
            unsigned lo = bs_base + (unsigned)(r * BN + cc) * 2u;
            asm volatile("global_load_async_to_lds_b128 %0, %1, off"
                         :: "v"(lo), "v"(ga) : "memory");
        }
        asm volatile("s_wait_asynccnt 0x0" ::: "memory");
        __syncthreads();

        // ---- fragment assembly (contiguous k = 16*half + s, same in A and B) ----
        v16bf af[2], bfr[4];
        #pragma unroll
        for (int t = 0; t < 2; ++t) {
            int row = wm + t * 16 + l16;               // A: row = lane%16
            #pragma unroll
            for (int p = 0; p < 8; ++p) {
                int kk = 2 * p + half * 16;
                af[t][2 * p]     = As[row][kk];
                af[t][2 * p + 1] = As[row][kk + 1];
            }
        }
        #pragma unroll
        for (int t = 0; t < 4; ++t) {
            int col = wn + t * 16 + l16;               // B: col = lane%16
            #pragma unroll
            for (int p = 0; p < 8; ++p) {
                int kk = 2 * p + half * 16;
                bfr[t][2 * p]     = Bs[kk][col];
                bfr[t][2 * p + 1] = Bs[kk + 1][col];
            }
        }
        #pragma unroll
        for (int i = 0; i < 2; ++i)
            #pragma unroll
            for (int j = 0; j < 4; ++j)
                acc[i][j] = wmma_bf16(af[i], bfr[j], acc[i][j]);
        __syncthreads();
    }

    // Epilogue: C layout — VGPR e: row e (lanes 0-15) / row 8+e (lanes 16-31).
    #pragma unroll
    for (int i = 0; i < 2; ++i)
        #pragma unroll
        for (int j = 0; j < 4; ++j)
            #pragma unroll
            for (int e = 0; e < 8; ++e) {
                int r = m0 + wm + i * 16 + e + half * 8;
                int c = n0 + wn + j * 16 + l16;
                float v = acc[i][j][e] + bias[c];
                if (OUT_F32)
                    ((float*)Cptr)[(size_t)r * N + c] = v;
                else
                    ((__bf16*)Cptr)[(size_t)r * N + c] = (__bf16)v;
            }
}

// ---------------------------------------------------------------------------
// RoPE (in-place on bf16 buffer laid out [T][nheads*HD]).
// ---------------------------------------------------------------------------
__global__ __launch_bounds__(256)
void rope_kernel(__bf16* __restrict__ q, const float* __restrict__ freqs,
                 int total, int nheads)
{
    int idx = blockIdx.x * blockDim.x + threadIdx.x;
    if (idx >= total) return;
    const int halfd = HD / 2;
    const int per_t = nheads * halfd;
    int t   = idx / per_t;
    int rem = idx - t * per_t;
    int h   = rem / halfd;
    int i   = rem - h * halfd;
    float f = freqs[t * halfd + i];
    float c = cosf(f), s = sinf(f);
    size_t base = (size_t)t * nheads * HD + (size_t)h * HD;
    float x1 = (float)q[base + i];
    float x2 = (float)q[base + i + halfd];
    q[base + i]         = (__bf16)(x1 * c - x2 * s);
    q[base + i + halfd] = (__bf16)(x1 * s + x2 * c);
}

// ---------------------------------------------------------------------------
// Flash attention: one wave (32 threads) per (16-query tile, head).
// ---------------------------------------------------------------------------
__global__ __launch_bounds__(32)
void flash_attn(const __bf16* __restrict__ Q, const __bf16* __restrict__ Kb,
                const __bf16* __restrict__ Vb, __bf16* __restrict__ O)
{
    const int qblk = blockIdx.x;
    const int h    = blockIdx.y;
    const int kvh  = h >> 2;              // GQA 4:1
    const int lane = threadIdx.x;
    const int half = lane >> 4;
    const int l16  = lane & 15;
    const int q0   = qblk * 16;

    __shared__ __bf16 p_lds[16][32];      // P transpose bounce (C -> A layout)

    v16bf aq[2];
    {
        const __bf16* qrow = Q + (size_t)(q0 + l16) * (NQH * HD) + (size_t)h * HD;
        #pragma unroll
        for (int s = 0; s < 2; ++s)
            #pragma unroll
            for (int p = 0; p < 8; ++p) {
                int kk = 2 * p + half * 16 + s * 32;
                aq[s][2 * p]     = qrow[kk];
                aq[s][2 * p + 1] = qrow[kk + 1];
            }
    }

    v8f oacc[4];
    #pragma unroll
    for (int c = 0; c < 4; ++c) oacc[c] = (v8f)0.0f;
    float m_i[8], l_i[8];
    #pragma unroll
    for (int e = 0; e < 8; ++e) { m_i[e] = -__builtin_inff(); l_i[e] = 0.0f; }

    const int jend = (q0 + 15) / 32;
    for (int jb = 0; jb <= jend; ++jb) {
        const int kbase = jb * 32;

        v8f s[2];
        #pragma unroll
        for (int ht = 0; ht < 2; ++ht) {
            v16bf bk0, bk1;
            const __bf16* krow =
                Kb + (size_t)(kbase + ht * 16 + l16) * (NKVH * HD) + (size_t)kvh * HD;
            #pragma unroll
            for (int p = 0; p < 8; ++p) {
                int dd = 2 * p + half * 16;
                bk0[2 * p]     = krow[dd];
                bk0[2 * p + 1] = krow[dd + 1];
                bk1[2 * p]     = krow[dd + 32];
                bk1[2 * p + 1] = krow[dd + 33];
            }
            v8f c = (v8f)0.0f;
            c = wmma_bf16(aq[0], bk0, c);
            c = wmma_bf16(aq[1], bk1, c);
            s[ht] = c;
        }

        float pm[2][8];
        #pragma unroll
        for (int e = 0; e < 8; ++e) {
            const int trow = q0 + e + half * 8;
            #pragma unroll
            for (int ht = 0; ht < 2; ++ht) {
                int key = kbase + ht * 16 + l16;
                float v = s[ht][e] * 0.125f;              // 1/sqrt(64)
                pm[ht][e] = (key > trow) ? -__builtin_inff() : v;
            }
            float cand = fmaxf(pm[0][e], pm[1][e]);
            #pragma unroll
            for (int off = 8; off > 0; off >>= 1)
                cand = fmaxf(cand, __shfl_xor(cand, off, 32));
            float mnew = fmaxf(m_i[e], cand);
            float fac  = exp2f((m_i[e] - mnew) * LOG2E);
            float rs   = 0.0f;
            #pragma unroll
            for (int ht = 0; ht < 2; ++ht) {
                float p = exp2f((pm[ht][e] - mnew) * LOG2E);
                pm[ht][e] = p;
                rs += p;
            }
            #pragma unroll
            for (int off = 8; off > 0; off >>= 1)
                rs += __shfl_xor(rs, off, 32);
            l_i[e] = l_i[e] * fac + rs;
            m_i[e] = mnew;
            #pragma unroll
            for (int c = 0; c < 4; ++c) oacc[c][e] *= fac;
        }

        #pragma unroll
        for (int e = 0; e < 8; ++e) {
            p_lds[e + half * 8][l16]      = (__bf16)pm[0][e];
            p_lds[e + half * 8][16 + l16] = (__bf16)pm[1][e];
        }
        __syncthreads();
        v16bf ap;
        #pragma unroll
        for (int p = 0; p < 8; ++p) {
            int kk = 2 * p + half * 16;
            ap[2 * p]     = p_lds[l16][kk];
            ap[2 * p + 1] = p_lds[l16][kk + 1];
        }

        #pragma unroll
        for (int c = 0; c < 4; ++c) {
            v16bf bv;
            #pragma unroll
            for (int p = 0; p < 8; ++p) {
                int key = kbase + 2 * p + half * 16;
                const __bf16* vr =
                    Vb + (size_t)key * (NKVH * HD) + (size_t)kvh * HD + c * 16 + l16;
                bv[2 * p]     = vr[0];
                bv[2 * p + 1] = vr[NKVH * HD];   // key+1
            }
            oacc[c] = wmma_bf16(ap, bv, oacc[c]);
        }
        __syncthreads();
    }

    #pragma unroll
    for (int c = 0; c < 4; ++c)
        #pragma unroll
        for (int e = 0; e < 8; ++e) {
            int t = q0 + e + half * 8;
            O[(size_t)t * DMODEL + (size_t)h * HD + c * 16 + l16] =
                (__bf16)(oacc[c][e] / l_i[e]);
        }
}

// ---------------------------------------------------------------------------
// Launcher.  Inputs: x, Wq, bq, Wk, bk, Wv, bv, Wo, bo, freqs (all f32).
// Workspace layout (bf16):
//   Qb 8MB | Kb 2MB | Vb 2MB | Ab 8MB | xb 8MB | Wqb 8MB | Wkb 2MB | Wvb 2MB | Wob 8MB
// ---------------------------------------------------------------------------
extern "C" void kernel_launch(void* const* d_in, const int* /*in_sizes*/, int /*n_in*/,
                              void* d_out, int /*out_size*/, void* d_ws, size_t /*ws_size*/,
                              hipStream_t stream)
{
    const float* x     = (const float*)d_in[0];
    const float* Wq    = (const float*)d_in[1];
    const float* bq    = (const float*)d_in[2];
    const float* Wk    = (const float*)d_in[3];
    const float* bk    = (const float*)d_in[4];
    const float* Wv    = (const float*)d_in[5];
    const float* bv    = (const float*)d_in[6];
    const float* Wo    = (const float*)d_in[7];
    const float* bo    = (const float*)d_in[8];
    const float* freqs = (const float*)d_in[9];

    const size_t SZ_TD  = (size_t)T_CTX * DMODEL;       // 4M elems
    const size_t SZ_TKV = (size_t)T_CTX * (NKVH * HD);  // 1M elems
    const size_t SZ_DKV = (size_t)DMODEL * (NKVH * HD); // 1M elems

    __bf16* Qb  = (__bf16*)d_ws;
    __bf16* Kb  = Qb  + SZ_TD;
    __bf16* Vb  = Kb  + SZ_TKV;
    __bf16* Ab  = Vb  + SZ_TKV;
    __bf16* xb  = Ab  + SZ_TD;
    __bf16* Wqb = xb  + SZ_TD;
    __bf16* Wkb = Wqb + SZ_TD;
    __bf16* Wvb = Wkb + SZ_DKV;
    __bf16* Wob = Wvb + SZ_DKV;

    // One-time f32 -> bf16 conversions (x + all weights).
    auto cvt = [&](const float* s, __bf16* d, size_t n) {
        int n4 = (int)(n / 4);
        cvt_bf16_kernel<<<(n4 + 255) / 256, 256, 0, stream>>>(s, d, n4);
    };
    cvt(x,  xb,  SZ_TD);
    cvt(Wq, Wqb, SZ_TD);
    cvt(Wk, Wkb, SZ_DKV);
    cvt(Wv, Wvb, SZ_DKV);
    cvt(Wo, Wob, SZ_TD);

    dim3 blk(256);
    // QKV projections (bf16 x bf16 -> bf16 out, fused bias).
    gemm_bf16_wmma<false><<<dim3(DMODEL / 128, T_CTX / 128), blk, 0, stream>>>(
        xb, Wqb, bq, Qb, T_CTX, DMODEL, DMODEL);
    gemm_bf16_wmma<false><<<dim3((NKVH * HD) / 128, T_CTX / 128), blk, 0, stream>>>(
        xb, Wkb, bk, Kb, T_CTX, NKVH * HD, DMODEL);
    gemm_bf16_wmma<false><<<dim3((NKVH * HD) / 128, T_CTX / 128), blk, 0, stream>>>(
        xb, Wvb, bv, Vb, T_CTX, NKVH * HD, DMODEL);

    // RoPE on Q and K.
    int nQ = T_CTX * NQH * (HD / 2);
    int nK = T_CTX * NKVH * (HD / 2);
    rope_kernel<<<(nQ + 255) / 256, 256, 0, stream>>>(Qb, freqs, nQ, NQH);
    rope_kernel<<<(nK + 255) / 256, 256, 0, stream>>>(Kb, freqs, nK, NKVH);

    // Causal GQA flash attention.
    flash_attn<<<dim3(T_CTX / 16, NQH), 32, 0, stream>>>(Qb, Kb, Vb, Ab);

    // Output projection (bf16 in -> f32 out, fused bias).
    gemm_bf16_wmma<true><<<dim3(DMODEL / 128, T_CTX / 128), blk, 0, stream>>>(
        Ab, Wob, bo, (float*)d_out, T_CTX, DMODEL, DMODEL);
}
